// GNN_59219009077676
// MI455X (gfx1250) — matrix operand order repaired and checked by reading, not compiled, not run on hardware
//
#include <hip/hip_runtime.h>
#include <math.h>

typedef __attribute__((ext_vector_type(16))) _Float16 v16h;
typedef __attribute__((ext_vector_type(8)))  float    v8f;

#define NN   100000
#define EE   1600000
#define GG   1000
#define GP   1008    // G padded to multiple of 16 for guard-free small GEMMs
#define DIN  128
#define DD   300
#define LL   5
#define CC   10
#define P1   320     // padded 300-wide (10 K-chunks of 32)
#define P2   608     // padded 600-wide (19 K-chunks of 32)
#define KCMAX 19     // max K-chunks (608/32)

// ---------------------------------------------------------------------------
// WMMA GEMM:  C[Mp x ldc] = act((A[Mp x lda] @ B[K x Nn,ldb] + bias)*scale + shift)
// * A comes from padded workspace: lda % 32 == 0, Mp % 16 == 0, pad cols zero.
//   -> A fragment = 4 unconditional float4 loads per lane per 32-K chunk.
// * B (fp32 weights, unpadded) is staged once per block into LDS in f16
//   WMMA-fragment order (all bounds checks only in the one-time fill).
//   -> B fragment = one 32B LDS read per lane per chunk.
// * Inner loop is branch-free: global_load_b128 x4 + ds reads + cvt + v_wmma.
// ---------------------------------------------------------------------------
__global__ __launch_bounds__(128)
void wmma_gemm(const float* __restrict__ A, int lda,
               const float* __restrict__ B, int ldb,
               float* __restrict__ C, int ldc,
               int Mp, int K, int Nn,
               const float* __restrict__ bias,
               const float* __restrict__ scale,
               const float* __restrict__ shift,
               int relu, int kChunks) {
  __shared__ _Float16 Bs[KCMAX * 32 * 16];   // [kc][lane][16 halves] = 19456 B

  const int tileN = blockIdx.x;
  const int lane  = threadIdx.x;             // 0..31 (wave32)
  const int tid   = threadIdx.y * 32 + lane; // 0..127

  // ---- cooperative B panel fill (once per block) -------------------------
  for (int idx = tid; idx < kChunks * 256; idx += 128) {
    const int kc   = idx >> 8;
    const int rem  = idx & 255;
    const int v    = rem & 7;
    const int ln   = rem >> 3;               // fragment lane 0..31
    const int k    = kc * 32 + ((v >> 2) << 4) + ((ln >> 4) << 3) + ((v & 3) << 1);
    const int n    = tileN * 16 + (ln & 15);
    const bool cOK = n < Nn;
    const float b0 = (cOK && k     < K) ? B[(size_t)k * ldb + n]       : 0.0f;
    const float b1 = (cOK && k + 1 < K) ? B[(size_t)(k + 1) * ldb + n] : 0.0f;
    _Float16* d = &Bs[(kc * 32 + ln) * 16 + 2 * v];
    d[0] = (_Float16)b0;
    d[1] = (_Float16)b1;
  }
  __syncthreads();

  const int tileM = blockIdx.y * 4 + threadIdx.y;
  if (tileM * 16 >= Mp) return;

  const int mHalf = lane >> 4;               // 0 or 1
  const int mloc  = lane & 15;
  const int rowA  = tileM * 16 + mloc;       // always < Mp (buffers padded)
  const float4* __restrict__ Arow4 =
      (const float4*)(A + (size_t)rowA * lda);
  const _Float16* __restrict__ BsLane = &Bs[lane * 16];

  v8f acc = {};
  for (int kc = 0; kc < kChunks; ++kc) {
    // A: 16 consecutive-in-pairs floats -> 4 x float4, branch-free
    const int i0 = kc * 8 + mHalf * 2;       // (kb/4 + mHalf*2)
    const float4 f0 = Arow4[i0];
    const float4 f1 = Arow4[i0 + 1];
    const float4 f2 = Arow4[i0 + 4];
    const float4 f3 = Arow4[i0 + 5];
    v16h a;
    a[0]  = (_Float16)f0.x;  a[1]  = (_Float16)f0.y;
    a[2]  = (_Float16)f0.z;  a[3]  = (_Float16)f0.w;
    a[4]  = (_Float16)f1.x;  a[5]  = (_Float16)f1.y;
    a[6]  = (_Float16)f1.z;  a[7]  = (_Float16)f1.w;
    a[8]  = (_Float16)f2.x;  a[9]  = (_Float16)f2.y;
    a[10] = (_Float16)f2.z;  a[11] = (_Float16)f2.w;
    a[12] = (_Float16)f3.x;  a[13] = (_Float16)f3.y;
    a[14] = (_Float16)f3.z;  a[15] = (_Float16)f3.w;

    // B: prebuilt fragment, one 32B LDS read
    const v16h b = *(const v16h*)(BsLane + (size_t)kc * 32 * 16);

    acc = __builtin_amdgcn_wmma_f32_16x16x32_f16(
        false, a, false, b, (short)0, acc, false, false);
  }

  // Epilogue: value (M = r + 8*mHalf, N = mloc) lives in acc[r]
  const int col = tileN * 16 + mloc;
  const bool cReal = col < Nn;
  const float bb = cReal ? bias[col] : 0.0f;
  const float ss = (scale && cReal) ? scale[col] : 1.0f;
  const float tt = (shift && cReal) ? shift[col] : 0.0f;
#pragma unroll
  for (int r = 0; r < 8; ++r) {
    const int row = tileM * 16 + r + (mHalf << 3);
    float v = acc[r] + bb;
    v = v * ss + tt;
    if (relu) v = fmaxf(v, 0.0f);
    C[(size_t)row * ldc + col] = cReal ? v : 0.0f;
  }
}

// ---------------------------------------------------------------------------
// Elementwise / scatter helpers
// ---------------------------------------------------------------------------
__global__ void k_fill(float* p, float v, size_t n) {
  size_t i = (size_t)blockIdx.x * blockDim.x + threadIdx.x;
  if (i < n) p[i] = v;
}

__global__ void k_copy(float* __restrict__ d, const float* __restrict__ s, size_t n) {
  size_t i = (size_t)blockIdx.x * blockDim.x + threadIdx.x;
  if (i < n) d[i] = s[i];
}

__global__ void k_vn_init(float* vn, const float* __restrict__ vn0) {
  size_t i = (size_t)blockIdx.x * blockDim.x + threadIdx.x;
  if (i < (size_t)GP * P1) {
    int c = (int)(i % P1);
    vn[i] = (c < DD) ? vn0[c] : 0.0f;   // pad rows get vn0 too; never read
  }
}

// hl = h + vn[batch]   (full padded width; pads stay 0)
__global__ void k_hl(float* __restrict__ hl, const float* __restrict__ h,
                     const float* __restrict__ vn, const long long* __restrict__ batch) {
  size_t i = (size_t)blockIdx.x * blockDim.x + threadIdx.x;
  if (i < (size_t)NN * P1) {
    int row = (int)(i / P1);
    int col = (int)(i % P1);
    long long g = batch[row];
    hl[i] = h[i] + vn[(size_t)g * P1 + col];
  }
}

// agg = (1 + eps[l]) * hl
__global__ void k_selfscale(float* __restrict__ agg, const float* __restrict__ hl,
                            const float* __restrict__ eps, int l) {
  size_t i = (size_t)blockIdx.x * blockDim.x + threadIdx.x;
  if (i < (size_t)NN * P1) agg[i] = (1.0f + eps[l]) * hl[i];
}

// agg[dst] += hl[src]   grid.y = col-group (4 floats), grid.x covers edges
__global__ void k_scatter(float* __restrict__ agg, const float* __restrict__ hl,
                          const long long* __restrict__ ei) {
  long long e = (long long)blockIdx.x * blockDim.x + threadIdx.x;
  if (e >= EE) return;
  int c = blockIdx.y * 4;
  long long src = ei[e];
  long long dst = ei[(long long)EE + e];
  const float4 v = *(const float4*)(hl + (size_t)src * P1 + c);
  float* d = agg + (size_t)dst * P1 + c;
  atomicAdd(d + 0, v.x);
  atomicAdd(d + 1, v.y);
  atomicAdd(d + 2, v.z);
  atomicAdd(d + 3, v.w);
}

// vtmp[batch[i]] += hl[i]   (cols 0..299)
__global__ void k_pool(float* __restrict__ vtmp, const float* __restrict__ hl,
                       const long long* __restrict__ batch) {
  int i = blockIdx.x * blockDim.x + threadIdx.x;
  if (i >= NN) return;
  int c = blockIdx.y * 4;
  long long g = batch[i];
  const float4 v = *(const float4*)(hl + (size_t)i * P1 + c);
  float* d = vtmp + (size_t)g * P1 + c;
  atomicAdd(d + 0, v.x);
  atomicAdd(d + 1, v.y);
  atomicAdd(d + 2, v.z);
  atomicAdd(d + 3, v.w);
}

__global__ void k_maxup(float* __restrict__ nr, const float* __restrict__ h) {
  size_t i = (size_t)blockIdx.x * blockDim.x + threadIdx.x;
  if (i < (size_t)NN * P1) nr[i] = fmaxf(nr[i], h[i]);
}

// gate[r] = dot(u2[r,0:600], gW2) + gb2
__global__ void k_gatedot(float* __restrict__ gate, const float* __restrict__ u2,
                          const float* __restrict__ gW2, const float* __restrict__ gb2) {
  int r = blockIdx.x * blockDim.x + threadIdx.x;
  if (r >= NN) return;
  const float* row = u2 + (size_t)r * P2;
  float acc = 0.0f;
  for (int k = 0; k < 2 * DD; ++k) acc += row[k] * gW2[k];
  gate[r] = acc + gb2[0];
}

__device__ inline void atomicMaxF(float* addr, float val) {
  unsigned* ua = (unsigned*)addr;
  unsigned old = *ua;
  while (__uint_as_float(old) < val) {
    unsigned assumed = old;
    old = atomicCAS(ua, assumed, __float_as_uint(val));
    if (old == assumed) break;
  }
}

__global__ void k_segmax(float* __restrict__ gmax, const float* __restrict__ gate,
                         const long long* __restrict__ batch) {
  int i = blockIdx.x * blockDim.x + threadIdx.x;
  if (i >= NN) return;
  atomicMaxF(gmax + batch[i], gate[i]);
}

// gate[i] = exp(gate[i] - gmax[b]); denom[b] += gate[i]
__global__ void k_expden(float* __restrict__ gate, const float* __restrict__ gmax,
                         float* __restrict__ denom, const long long* __restrict__ batch) {
  int i = blockIdx.x * blockDim.x + threadIdx.x;
  if (i >= NN) return;
  long long g = batch[i];
  float e = expf(gate[i] - gmax[g]);
  gate[i] = e;
  atomicAdd(denom + g, e);
}

// grep[b] += nr[i] * (e_i / denom[b])
__global__ void k_wpool(float* __restrict__ grep, const float* __restrict__ nr,
                        const float* __restrict__ gate, const float* __restrict__ denom,
                        const long long* __restrict__ batch) {
  int i = blockIdx.x * blockDim.x + threadIdx.x;
  if (i >= NN) return;
  int c = blockIdx.y * 4;
  long long g = batch[i];
  float w = gate[i] / denom[g];
  const float4 v = *(const float4*)(nr + (size_t)i * P1 + c);
  float* d = grep + (size_t)g * P1 + c;
  atomicAdd(d + 0, v.x * w);
  atomicAdd(d + 1, v.y * w);
  atomicAdd(d + 2, v.z * w);
  atomicAdd(d + 3, v.w * w);
}

// out[g,c] = dot(grep[g,0:300], pW[:,c]) + pb[c]
__global__ void k_out(float* __restrict__ out, const float* __restrict__ grep,
                      const float* __restrict__ pW, const float* __restrict__ pb) {
  int t = blockIdx.x * blockDim.x + threadIdx.x;
  if (t >= GG * CC) return;
  int g = t / CC, c = t % CC;
  const float* row = grep + (size_t)g * P1;
  float acc = 0.0f;
  for (int k = 0; k < DD; ++k) acc += row[k] * pW[k * CC + c];
  out[t] = acc + pb[c];
}

// ---------------------------------------------------------------------------
// Host orchestration
// ---------------------------------------------------------------------------
static inline size_t ceil_div(size_t a, size_t b) { return (a + b - 1) / b; }

extern "C" void kernel_launch(void* const* d_in, const int* in_sizes, int n_in,
                              void* d_out, int out_size, void* d_ws, size_t ws_size,
                              hipStream_t stream) {
  const float*     x      = (const float*)d_in[0];
  const long long* ei     = (const long long*)d_in[1];
  const long long* batch  = (const long long*)d_in[2];
  const float* enc_W  = (const float*)d_in[3];
  const float* enc_b  = (const float*)d_in[4];
  const float* eps    = (const float*)d_in[5];
  const float* cW1    = (const float*)d_in[6];
  const float* cb1    = (const float*)d_in[7];
  const float* cbn_s  = (const float*)d_in[8];
  const float* cbn_b  = (const float*)d_in[9];
  const float* cW2    = (const float*)d_in[10];
  const float* cb2    = (const float*)d_in[11];
  const float* bn_s   = (const float*)d_in[12];
  const float* bn_b   = (const float*)d_in[13];
  const float* vn0    = (const float*)d_in[14];
  const float* vW1    = (const float*)d_in[15];
  const float* vb1    = (const float*)d_in[16];
  const float* vbn1_s = (const float*)d_in[17];
  const float* vbn1_b = (const float*)d_in[18];
  const float* vW2    = (const float*)d_in[19];
  const float* vb2    = (const float*)d_in[20];
  const float* vbn2_s = (const float*)d_in[21];
  const float* vbn2_b = (const float*)d_in[22];
  const float* gW1    = (const float*)d_in[23];
  const float* gb1    = (const float*)d_in[24];
  const float* gbn_s  = (const float*)d_in[25];
  const float* gbn_b  = (const float*)d_in[26];
  const float* gW2    = (const float*)d_in[27];
  const float* gb2    = (const float*)d_in[28];
  const float* pW     = (const float*)d_in[29];
  const float* pb     = (const float*)d_in[30];
  float* out = (float*)d_out;

  // workspace carve-up (floats)
  float* ws = (float*)d_ws;
  size_t o = 0;
  float* h    = ws + o; o += (size_t)NN * P1;
  float* hl   = ws + o; o += (size_t)NN * P1;
  float* agg  = ws + o; o += (size_t)NN * P1;
  float* nr   = ws + o; o += (size_t)NN * P1;
  float* t1   = ws + o; o += (size_t)NN * P2;
  float* vn   = ws + o; o += (size_t)GP * P1;
  float* vtmp = ws + o; o += (size_t)GP * P1;
  float* u    = ws + o; o += (size_t)GP * P2;
  float* gate = ws + o; o += (size_t)NN;
  float* gmax = ws + o; o += (size_t)GG;
  float* den  = ws + o; o += (size_t)GG;
  float* grep = ws + o; o += (size_t)GG * P1;
  (void)ws_size; (void)in_sizes; (void)n_in; (void)out_size;

  const int TB = 256;
  const size_t NP1 = (size_t)NN * P1;

  auto gemm = [&](const float* A, int lda, const float* B, int ldb,
                  float* Cp, int ldc, int Mp, int K, int Nn,
                  const float* bias, const float* scale, const float* shift,
                  int relu) {
    dim3 blk(32, 4);
    int mt = Mp / 16;                  // Mp always multiple of 16
    dim3 grd((unsigned)(ldc / 16), (unsigned)((mt + 3) / 4));
    wmma_gemm<<<grd, blk, 0, stream>>>(A, lda, B, ldb, Cp, ldc, Mp, K, Nn,
                                       bias, scale, shift, relu, lda / 32);
  };

  // --- encoder: h0 = x @ enc_W + enc_b -----------------------------------
  gemm(x, DIN, enc_W, DD, h, P1, NN, DIN, DD, enc_b, nullptr, nullptr, 0);

  // nr = h0
  k_copy<<<ceil_div(NP1, TB), TB, 0, stream>>>(nr, h, NP1);
  // vn = broadcast(vn0) over padded G rows
  k_vn_init<<<ceil_div((size_t)GP * P1, TB), TB, 0, stream>>>(vn, vn0);

  for (int l = 0; l < LL; ++l) {
    // hl = h + vn[batch]
    k_hl<<<ceil_div(NP1, TB), TB, 0, stream>>>(hl, h, vn, batch);
    // vtmp = vn ; vtmp[batch[i]] += hl[i]
    k_copy<<<ceil_div((size_t)GP * P1, TB), TB, 0, stream>>>(vtmp, vn, (size_t)GP * P1);
    {
      dim3 grd((unsigned)ceil_div(NN, TB), DD / 4);
      k_pool<<<grd, TB, 0, stream>>>(vtmp, hl, batch);
    }
    // agg = (1+eps[l])*hl ; agg[dst] += hl[src]
    k_selfscale<<<ceil_div(NP1, TB), TB, 0, stream>>>(agg, hl, eps, l);
    {
      dim3 grd((unsigned)ceil_div(EE, TB), DD / 4);
      k_scatter<<<grd, TB, 0, stream>>>(agg, hl, ei);
    }
    // conv MLP: t1 = relu((agg@cW1 + cb1)*cbn_s + cbn_b)
    gemm(agg, P1, cW1 + (size_t)l * DD * 2 * DD, 2 * DD, t1, P2, NN, DD, 2 * DD,
         cb1 + l * 2 * DD, cbn_s + l * 2 * DD, cbn_b + l * 2 * DD, 1);
    // h = (t1@cW2 + cb2)*bn_s + bn_b ; relu if l < L-1
    gemm(t1, P2, cW2 + (size_t)l * 2 * DD * DD, DD, h, P1, NN, 2 * DD, DD,
         cb2 + l * DD, bn_s + l * DD, bn_b + l * DD, (l < LL - 1) ? 1 : 0);
    // virtual node MLP (padded M = GP, pad rows are junk but never read)
    gemm(vtmp, P1, vW1 + (size_t)l * DD * 2 * DD, 2 * DD, u, P2, GP, DD, 2 * DD,
         vb1 + l * 2 * DD, vbn1_s + l * 2 * DD, vbn1_b + l * 2 * DD, 1);
    gemm(u, P2, vW2 + (size_t)l * 2 * DD * DD, DD, vn, P1, GP, 2 * DD, DD,
         vb2 + l * DD, vbn2_s + l * DD, vbn2_b + l * DD, 1);
    // nr = max(nr, h)
    k_maxup<<<ceil_div(NP1, TB), TB, 0, stream>>>(nr, h);
  }

  // --- global attention pooling ------------------------------------------
  gemm(nr, P1, gW1, 2 * DD, t1, P2, NN, DD, 2 * DD, gb1, gbn_s, gbn_b, 1);
  k_gatedot<<<ceil_div(NN, TB), TB, 0, stream>>>(gate, t1, gW2, gb2);

  k_fill<<<ceil_div(GG, TB), TB, 0, stream>>>(gmax, -3.0e38f, GG);
  hipMemsetAsync(den, 0, GG * sizeof(float), stream);
  hipMemsetAsync(grep, 0, (size_t)GG * P1 * sizeof(float), stream);

  k_segmax<<<ceil_div(NN, TB), TB, 0, stream>>>(gmax, gate, batch);
  k_expden<<<ceil_div(NN, TB), TB, 0, stream>>>(gate, gmax, den, batch);
  {
    dim3 grd((unsigned)ceil_div(NN, TB), DD / 4);
    k_wpool<<<grd, TB, 0, stream>>>(grep, nr, gate, den, batch);
  }
  k_out<<<ceil_div(GG * CC, TB), TB, 0, stream>>>(out, grep, pW, pb);
}
